// MultiHeadAttention_62723702391184
// MI455X (gfx1250) — matrix operand order repaired and checked by reading, not compiled, not run on hardware
//
#include <hip/hip_runtime.h>

#define DD 1024
#define BB 2
#define SS 2048
#define HH 16
#define HD 64
#define MM (BB * SS) // 4096

typedef __attribute__((ext_vector_type(16))) _Float16 v16h;
typedef __attribute__((ext_vector_type(8)))  _Float16 v8h;
typedef __attribute__((ext_vector_type(8)))  float    v8f;
typedef __attribute__((ext_vector_type(4)))  float    v4f;

__device__ __forceinline__ v8f wmma16(v16h a, v16h b, v8f c) {
  // D = A(16x32 f16) * B(32x16 f16) + C(16x16 f32)
  return __builtin_amdgcn_wmma_f32_16x16x32_f16(false, a, false, b, (short)0, c,
                                                false, false);
}

// Load a 16x32 f16 A-fragment. `row32` points at the 32 contiguous K values of
// this lane's row (m = lane % 16). Contiguous order: element s holds
// K = (lane/16)*16 + s, matching the B operand loads.
__device__ __forceinline__ v16h load_a_frag(const _Float16* row32, int lane) {
  const int g = (lane >> 4) << 4;
  v8h lo = *(const v8h*)(row32 + g);
  v8h hi = *(const v8h*)(row32 + 8 + g);
  v16h r;
#pragma unroll
  for (int i = 0; i < 8; ++i) { r[i] = lo[i]; r[8 + i] = hi[i]; }
  return r;
}

// ---------------------------------------------------------------------------
// Weight transpose + fp32 -> f16 convert: Wt[n*D + k] = (f16) W[k*D + n]
// ---------------------------------------------------------------------------
__global__ __launch_bounds__(256) void wconv_kernel(const float* __restrict__ W,
                                                    _Float16* __restrict__ Wt) {
  const int idx = blockIdx.x * 256 + threadIdx.x; // over D*D, coalesced read
  const int k = idx >> 10;
  const int n = idx & 1023;
  Wt[(size_t)n * DD + k] = (_Float16)W[idx];
}

// ---------------------------------------------------------------------------
// Projection GEMM:  Out = X(fp32)[4096xD] * Wt^T + bias, stored f16.
// vmode 0: [B,H,S,HD] (Q,K)    vmode 1: [B,H,HD,S] (V transposed)
// Block = 8 waves, block tile 64M x 128N, wave tile 16M x 64N.
// K-loop is software pipelined with ping-pong register buffers.
// ---------------------------------------------------------------------------
__global__ __launch_bounds__(256) void gemm_proj_kernel(
    const float* __restrict__ X, const _Float16* __restrict__ Wt,
    const float* __restrict__ bias, _Float16* __restrict__ Out, int vmode) {
  const int lane = threadIdx.x & 31;
  const int w    = threadIdx.x >> 5;
  const int mw = w & 3, nw = w >> 2;
  const int m0 = blockIdx.x * 64 + mw * 16;
  const int n0 = blockIdx.y * 128 + nw * 64;
  const int lr = lane & 15, lh = lane >> 4;

  v8f acc[4] = {};
  const float* xrow = X + (size_t)(m0 + lr) * DD;

  auto load_x = [&](int kk, v4f* x) {
    const float* xp = xrow + kk + lh * 16;
    x[0] = *(const v4f*)(xp);
    x[1] = *(const v4f*)(xp + 4);
    x[2] = *(const v4f*)(xp + 8);
    x[3] = *(const v4f*)(xp + 12);
  };
  auto load_b = [&](int kk, v16h* b) {
#pragma unroll
    for (int nn = 0; nn < 4; ++nn)
      b[nn] = *(const v16h*)(Wt + (size_t)(n0 + nn * 16 + lr) * DD + kk +
                             (lh << 4));
  };
  auto compute = [&](const v4f* x, const v16h* b) {
    v16h a;
#pragma unroll
    for (int i = 0; i < 4; ++i) {
      a[i]      = (_Float16)x[0][i];
      a[4 + i]  = (_Float16)x[1][i];
      a[8 + i]  = (_Float16)x[2][i];
      a[12 + i] = (_Float16)x[3][i];
    }
#pragma unroll
    for (int nn = 0; nn < 4; ++nn) acc[nn] = wmma16(a, b[nn], acc[nn]);
  };

  v4f x0[4], x1[4];
  v16h b0[4], b1[4];
  load_x(0, x0);
  load_b(0, b0);
#pragma unroll 1
  for (int kk = 0; kk < DD; kk += 64) {
    load_x(kk + 32, x1);      // issue next tile's loads
    load_b(kk + 32, b1);
    compute(x0, b0);          // WMMAs overlap the loads above
    const int kn = (kk + 64 < DD) ? kk + 64 : 0; // harmless dummy on last iter
    load_x(kn, x0);
    load_b(kn, b0);
    compute(x1, b1);
  }

#pragma unroll
  for (int nn = 0; nn < 4; ++nn) {
    const int col = n0 + nn * 16 + lr;
    const float bv = bias[col];
    const int h = col >> 6, hd = col & 63;
#pragma unroll
    for (int r = 0; r < 8; ++r) {
      const int m = m0 + r + (lh << 3); // C layout: VGPR r, lane-half lh
      const int b = m >> 11;            // m / S
      const int s = m & (SS - 1);
      const float v = acc[nn][r] + bv;
      size_t idx;
      if (vmode) idx = ((size_t)(b * HH + h) * HD + hd) * SS + s;
      else       idx = ((size_t)(b * HH + h) * SS + s) * HD + hd;
      Out[idx] = (_Float16)v;
    }
  }
}

// ---------------------------------------------------------------------------
// Flash attention. grid = (S/128, B*H), block = 8 waves.
// Each wave owns a 16-row Q block, streams keys in tiles of 32 with online
// softmax. Pipelined: V-frags issued before the score WMMAs, next K-frags
// issued right after them; softmax VALU overlaps both. P is reshaped
// C-layout -> A-layout through a private per-wave LDS tile.
// ---------------------------------------------------------------------------
__global__ __launch_bounds__(256) void attn_kernel(
    const _Float16* __restrict__ Q, const _Float16* __restrict__ K,
    const _Float16* __restrict__ Vt, _Float16* __restrict__ AO) {
  __shared__ __align__(32) _Float16 Plds[8][16][32];

  const int lane = threadIdx.x & 31;
  const int w = threadIdx.x >> 5;
  const int lr = lane & 15, lh = lane >> 4;
  const int bh = blockIdx.y;
  const int bidx = bh >> 4, h = bh & 15;
  const int m0 = blockIdx.x * 128 + w * 16;

  const _Float16* Qh = Q + (size_t)bh * SS * HD;
  const _Float16* Kh = K + (size_t)bh * SS * HD;
  const _Float16* Vh = Vt + (size_t)bh * HD * SS;

  v16h qf[2];
#pragma unroll
  for (int c = 0; c < 2; ++c)
    qf[c] = load_a_frag(Qh + (size_t)(m0 + lr) * HD + c * 32, lane);

  float rmax[8], rsum[8];
#pragma unroll
  for (int r = 0; r < 8; ++r) { rmax[r] = -3.0e38f; rsum[r] = 0.0f; }
  v8f oacc[4] = {};

  _Float16* prow = &Plds[w][lr][0];

  // kf[0]=keys[0:16) hd[0:32), kf[1]=keys[16:32) hd[0:32),
  // kf[2]=keys[0:16) hd[32:64), kf[3]=keys[16:32) hd[32:64)
  auto load_k = [&](int kt, v16h* kf) {
    const _Float16* k0p = Kh + (size_t)(kt + lr) * HD + (lh << 4);
#pragma unroll
    for (int c = 0; c < 2; ++c) {
      kf[2 * c]     = *(const v16h*)(k0p + c * 32);
      kf[2 * c + 1] = *(const v16h*)(k0p + 16 * HD + c * 32);
    }
  };

  v16h kf[4];
  load_k(0, kf);

#pragma unroll 1
  for (int kt = 0; kt < SS; kt += 32) {
    // ---- issue V loads for this tile early ----
    v16h vf[4];
#pragma unroll
    for (int nn = 0; nn < 4; ++nn)
      vf[nn] = *(const v16h*)(Vh + (size_t)(nn * 16 + lr) * SS + kt + (lh << 4));

    // ---- scores: S = Q(16x64) * K^T(64x32), two 16x16 tiles ----
    v8f sc0 = {}, sc1 = {};
    sc0 = wmma16(qf[0], kf[0], sc0);
    sc1 = wmma16(qf[0], kf[1], sc1);
    sc0 = wmma16(qf[1], kf[2], sc0);
    sc1 = wmma16(qf[1], kf[3], sc1);

    // ---- issue next K tile's loads; prefetch further ahead ----
    const int kn = (kt + 32 < SS) ? kt + 32 : 0; // harmless dummy on last iter
    load_k(kn, kf);
    if (kt + 128 < SS) {
      __builtin_prefetch(Kh + (size_t)(kt + 128 + lr) * HD, 0, 1);
      __builtin_prefetch(Vh + (size_t)lr * SS + kt + 128, 0, 1);
    }

    // ---- online softmax (row stats per (reg, lane-half); 16-lane reduce) ----
#pragma unroll
    for (int r = 0; r < 8; ++r) {
      float s0 = sc0[r] * 0.125f; // 1/sqrt(64)
      float s1 = sc1[r] * 0.125f;
      float rowm = fmaxf(s0, s1);
#pragma unroll
      for (int off = 1; off < 16; off <<= 1)
        rowm = fmaxf(rowm, __shfl_xor(rowm, off, 32));
      const float mn = fmaxf(rmax[r], rowm);
      const float f = __expf(rmax[r] - mn);
      const float p0 = __expf(s0 - mn);
      const float p1 = __expf(s1 - mn);
      rmax[r] = mn;
      rsum[r] = rsum[r] * f + p0 + p1;
#pragma unroll
      for (int nn = 0; nn < 4; ++nn) oacc[nn][r] *= f;
      Plds[w][r + (lh << 3)][lr]      = (_Float16)p0;
      Plds[w][r + (lh << 3)][16 + lr] = (_Float16)p1;
    }
    asm volatile("s_wait_dscnt 0" ::: "memory");

    // ---- P(16x32) * V(32x64): reload P in A layout, V^T rows contiguous ----
    v16h pa = load_a_frag(prow, lane);
    oacc[0] = wmma16(pa, vf[0], oacc[0]);
    oacc[1] = wmma16(pa, vf[1], oacc[1]);
    oacc[2] = wmma16(pa, vf[2], oacc[2]);
    oacc[3] = wmma16(pa, vf[3], oacc[3]);
  }

  // final row-sum reduction + normalize + store [B,S,D] f16
#pragma unroll
  for (int r = 0; r < 8; ++r) {
    float s = rsum[r];
#pragma unroll
    for (int off = 1; off < 16; off <<= 1) s += __shfl_xor(s, off, 32);
    rsum[r] = 1.0f / s;
  }
#pragma unroll
  for (int nn = 0; nn < 4; ++nn) {
#pragma unroll
    for (int r = 0; r < 8; ++r) {
      const int m = m0 + r + (lh << 3);
      const int col = h * HD + nn * 16 + lr;
      AO[(size_t)(bidx * SS + m) * DD + col] = (_Float16)(oacc[nn][r] * rsum[r]);
    }
  }
}

// ---------------------------------------------------------------------------
// Output projection: Out(fp32) = AO(f16) * Wo^T + bo, software pipelined.
// ---------------------------------------------------------------------------
__global__ __launch_bounds__(256) void gemm_out_kernel(
    const _Float16* __restrict__ A, const _Float16* __restrict__ Wt,
    const float* __restrict__ bias, float* __restrict__ Out) {
  const int lane = threadIdx.x & 31;
  const int w = threadIdx.x >> 5;
  const int mw = w & 3, nw = w >> 2;
  const int m0 = blockIdx.x * 64 + mw * 16;
  const int n0 = blockIdx.y * 128 + nw * 64;
  const int lr = lane & 15, lh = lane >> 4;

  v8f acc[4] = {};
  const _Float16* arow = A + (size_t)(m0 + lr) * DD;

  auto load_b = [&](int kk, v16h* b) {
#pragma unroll
    for (int nn = 0; nn < 4; ++nn)
      b[nn] = *(const v16h*)(Wt + (size_t)(n0 + nn * 16 + lr) * DD + kk +
                             (lh << 4));
  };
  auto compute = [&](v16h a, const v16h* b) {
#pragma unroll
    for (int nn = 0; nn < 4; ++nn) acc[nn] = wmma16(a, b[nn], acc[nn]);
  };

  v16h a0, a1, b0[4], b1[4];
  a0 = load_a_frag(arow, lane);
  load_b(0, b0);
#pragma unroll 1
  for (int kk = 0; kk < DD; kk += 64) {
    a1 = load_a_frag(arow + kk + 32, lane); // issue next tile's loads
    load_b(kk + 32, b1);
    compute(a0, b0);
    const int kn = (kk + 64 < DD) ? kk + 64 : 0; // dummy on last iter
    a0 = load_a_frag(arow + kn, lane);
    load_b(kn, b0);
    compute(a1, b1);
  }

#pragma unroll
  for (int nn = 0; nn < 4; ++nn) {
    const int col = n0 + nn * 16 + lr;
    const float bv = bias[col];
#pragma unroll
    for (int r = 0; r < 8; ++r) {
      const int m = m0 + r + (lh << 3);
      Out[(size_t)m * DD + col] = acc[nn][r] + bv;
    }
  }
}

// ---------------------------------------------------------------------------
extern "C" void kernel_launch(void* const* d_in, const int* in_sizes, int n_in,
                              void* d_out, int out_size, void* d_ws,
                              size_t ws_size, hipStream_t stream) {
  (void)in_sizes; (void)n_in; (void)out_size; (void)ws_size;
  const float* q  = (const float*)d_in[0];
  const float* k  = (const float*)d_in[1];
  const float* v  = (const float*)d_in[2];
  const float* Wq = (const float*)d_in[3];
  const float* bq = (const float*)d_in[4];
  const float* Wk = (const float*)d_in[5];
  const float* bk = (const float*)d_in[6];
  const float* Wv = (const float*)d_in[7];
  const float* bv = (const float*)d_in[8];
  const float* Wo = (const float*)d_in[9];
  const float* bo = (const float*)d_in[10];
  float* out = (float*)d_out;

  char* ws = (char*)d_ws;
  const size_t WSZ = (size_t)DD * DD * sizeof(_Float16); // 2 MB per weight
  const size_t TSZ = (size_t)MM * DD * sizeof(_Float16); // 8 MB per tensor
  _Float16* Wqt = (_Float16*)(ws);
  _Float16* Wkt = (_Float16*)(ws + WSZ);
  _Float16* Wvt = (_Float16*)(ws + 2 * WSZ);
  _Float16* Wot = (_Float16*)(ws + 3 * WSZ);
  _Float16* Qh  = (_Float16*)(ws + 4 * WSZ);
  _Float16* Kh  = (_Float16*)(ws + 4 * WSZ + TSZ);
  _Float16* Vt  = (_Float16*)(ws + 4 * WSZ + 2 * TSZ);
  _Float16* AO  = (_Float16*)(ws + 4 * WSZ + 3 * TSZ);

  dim3 blk(256);
  wconv_kernel<<<DD * DD / 256, blk, 0, stream>>>(Wq, Wqt);
  wconv_kernel<<<DD * DD / 256, blk, 0, stream>>>(Wk, Wkt);
  wconv_kernel<<<DD * DD / 256, blk, 0, stream>>>(Wv, Wvt);
  wconv_kernel<<<DD * DD / 256, blk, 0, stream>>>(Wo, Wot);

  dim3 gg(MM / 64, DD / 128);
  gemm_proj_kernel<<<gg, blk, 0, stream>>>(q, Wqt, bq, Qh, 0);
  gemm_proj_kernel<<<gg, blk, 0, stream>>>(k, Wkt, bk, Kh, 0);
  gemm_proj_kernel<<<gg, blk, 0, stream>>>(v, Wvt, bv, Vt, 1);

  attn_kernel<<<dim3(SS / 128, BB * HH), blk, 0, stream>>>(Qh, Kh, Vt, AO);

  gemm_out_kernel<<<gg, blk, 0, stream>>>(AO, Wot, bo, out);
}
